// SharedGNNBackbone_62723702391680
// MI455X (gfx1250) — compile-verified, matrix operand order checked by reading, not run on hardware
//
#include <hip/hip_runtime.h>
#include <hip/hip_bf16.h>

#define NN 40000
#define EE 640000
#define ET (EE + NN)   // 680000 edges incl. self loops

typedef __attribute__((ext_vector_type(16))) __bf16 v16bf;
typedef __attribute__((ext_vector_type(8)))  float  v8f;
typedef __attribute__((address_space(3))) unsigned short lds_us;

// ---------- helpers ----------
__device__ __forceinline__ unsigned short f2bf(float f) {
  unsigned u = __float_as_uint(f);
  unsigned r = u + 0x7FFFu + ((u >> 16) & 1u);   // RNE
  return (unsigned short)(r >> 16);
}
// order-preserving float<->uint key for atomicMax on floats (handles negatives)
__device__ __forceinline__ unsigned fkey(float f) {
  unsigned u = __float_as_uint(f);
  return (u & 0x80000000u) ? ~u : (u | 0x80000000u);
}
__device__ __forceinline__ float funkey(unsigned k) {
  return (k & 0x80000000u) ? __uint_as_float(k & 0x7FFFFFFFu) : __uint_as_float(~k);
}

// gfx1250 async global->LDS copy (ASYNCcnt path); fallback = direct copy
__device__ __forceinline__ void async_copy_b128(unsigned short* lp, const unsigned short* gp) {
#if defined(__AMDGCN__) && defined(__gfx1250__)
  unsigned loff = (unsigned)(uintptr_t)(lds_us*)lp;
  asm volatile("global_load_async_to_lds_b128 %0, %1, off"
               :: "v"(loff), "v"(gp) : "memory");
#else
  *(uint4*)lp = *(const uint4*)gp;
#endif
}
__device__ __forceinline__ void wait_async() {
#if defined(__AMDGCN__) && defined(__gfx1250__)
  asm volatile("s_wait_asynccnt 0x0" ::: "memory");
#endif
}

// ---------- elementwise / setup kernels ----------
__global__ void fill_u32_kernel(unsigned* __restrict__ p, unsigned v, long long n) {
  long long i = blockIdx.x * (long long)blockDim.x + threadIdx.x;
  if (i < n) p[i] = v;
}

__global__ void f32_to_bf16_kernel(const float* __restrict__ src,
                                   unsigned short* __restrict__ dst, long long n) {
  long long i = blockIdx.x * (long long)blockDim.x + threadIdx.x;
  if (i < n) dst[i] = f2bf(src[i]);
}

// Pack row-major weights [K, ncA(+ncB)] into WMMA bf16 B-fragment layout.
// Fragment (kt,nt): 32 lanes x 16 elems contiguous. Lane L: col = nt*16 + (L&15),
// kq = L>>4; elems 0..7 -> K = kt*32 + kq*8 + e; elems 8..15 -> K = kt*32 + 16 + kq*8 + (e-8).
__global__ void pack_b_kernel(const float* __restrict__ WA, const float* __restrict__ WB,
                              int ncA, int ncB, unsigned short* __restrict__ Bp, int Nc) {
  int t = blockIdx.x * blockDim.x + threadIdx.x;  // over K*Nc exactly
  int frag = t >> 9;
  int r = t & 511;
  int lane = r >> 4;
  int e = r & 15;
  int NT = Nc >> 4;
  int kt = frag / NT, nt = frag - kt * NT;
  int kq = lane >> 4;
  int kl = (e < 8) ? (kq * 8 + e) : (16 + kq * 8 + (e - 8));
  int k = kt * 32 + kl;
  int n = nt * 16 + (lane & 15);
  float v = (n < ncA) ? WA[(size_t)k * ncA + n] : WB[(size_t)k * ncB + (n - ncA)];
  Bp[t] = f2bf(v);
}

// ---------- WMMA GEMM: C[M,Nc] = A[M,K](bf16,row-major) @ Bpacked(bf16) ----------
// Block = 4 waves; each wave computes a 16x64 tile (4 accumulators) sharing one
// LDS-staged B working set (async global->LDS). M mult of 64, Nc mult of 64.
template <int K>
__global__ __launch_bounds__(128)
void gemm_bf16_wmma_kernel(const unsigned short* __restrict__ A,
                           const unsigned short* __restrict__ Bp,
                           float* __restrict__ Cmat, int Nc) {
  constexpr int KT = K >> 5;
  __shared__ __align__(16) unsigned short Bsh[K * 64];  // KT * 2048 elems

  const int lane  = threadIdx.x & 31;
  const int wave  = threadIdx.x >> 5;
  const int mtile = blockIdx.x * 4 + wave;
  const int nt0   = blockIdx.y * 4;
  const int row   = mtile * 16 + (lane & 15);
  const int kq    = lane >> 4;
  const int NT    = Nc >> 4;

  // stage all B fragments this block needs: KT chunks of 4 frags (2048 elems) each
#pragma unroll
  for (int kt = 0; kt < KT; ++kt) {
    const unsigned short* chunk = Bp + ((size_t)kt * NT + nt0) * 512;
#pragma unroll
    for (int r = 0; r < 2; ++r) {
      int off = (r * 128 + threadIdx.x) * 8;   // 8 elems = 16 bytes
      async_copy_b128(&Bsh[kt * 2048 + off], chunk + off);
    }
  }
  wait_async();
  __syncthreads();

  v8f acc0 = {}, acc1 = {}, acc2 = {}, acc3 = {};
#pragma unroll
  for (int kt = 0; kt < KT; ++kt) {
    union { v16bf v; uint4 q[2]; } af;
    const unsigned short* ap = A + (size_t)row * K + kt * 32 + kq * 8;
    af.q[0] = *(const uint4*)(ap);
    af.q[1] = *(const uint4*)(ap + 16);
    const unsigned short* bb = &Bsh[kt * 2048 + lane * 16];
    union { v16bf v; uint4 q[2]; } b0, b1, b2, b3;
    b0.q[0] = *(const uint4*)(bb);          b0.q[1] = *(const uint4*)(bb + 8);
    b1.q[0] = *(const uint4*)(bb + 512);    b1.q[1] = *(const uint4*)(bb + 520);
    b2.q[0] = *(const uint4*)(bb + 1024);   b2.q[1] = *(const uint4*)(bb + 1032);
    b3.q[0] = *(const uint4*)(bb + 1536);   b3.q[1] = *(const uint4*)(bb + 1544);
    acc0 = __builtin_amdgcn_wmma_f32_16x16x32_bf16(false, af.v, false, b0.v, (short)0, acc0, false, false);
    acc1 = __builtin_amdgcn_wmma_f32_16x16x32_bf16(false, af.v, false, b1.v, (short)0, acc1, false, false);
    acc2 = __builtin_amdgcn_wmma_f32_16x16x32_bf16(false, af.v, false, b2.v, (short)0, acc2, false, false);
    acc3 = __builtin_amdgcn_wmma_f32_16x16x32_bf16(false, af.v, false, b3.v, (short)0, acc3, false, false);
  }
  // C layout: lane -> col = lane&15 ; vgpr v -> row = v + 8*(lane>>4)
  const int crow0 = mtile * 16 + 8 * kq;
  const int ccol  = lane & 15;
  float* cp = Cmat + (size_t)crow0 * Nc + nt0 * 16 + ccol;
#pragma unroll
  for (int v = 0; v < 8; ++v) {
    cp[(size_t)v * Nc + 0]  = acc0[v];
    cp[(size_t)v * Nc + 16] = acc1[v];
    cp[(size_t)v * Nc + 32] = acc2[v];
    cp[(size_t)v * Nc + 48] = acc3[v];
  }
}

// ---------- attention scores: one wave per (node, head) ----------
__global__ void att_scores_kernel(const float* __restrict__ xp, int ld,
                                  const float* __restrict__ att_s,
                                  const float* __restrict__ att_d,
                                  float* __restrict__ as, float* __restrict__ ad,
                                  int pairs, int H, int C) {
  int gt = blockIdx.x * blockDim.x + threadIdx.x;
  int wid = gt >> 5, lane = gt & 31;
  if (wid >= pairs) return;
  int n = wid / H, h = wid - n * H;
  const float* xr = xp + (size_t)n * ld + h * C;
  const float* vs = att_s + h * C;
  const float* vd = att_d + h * C;
  float ss = 0.f, sd = 0.f;
  for (int c = lane; c < C; c += 32) { float xv = xr[c]; ss += xv * vs[c]; sd += xv * vd[c]; }
#pragma unroll
  for (int o = 16; o > 0; o >>= 1) { ss += __shfl_down(ss, o, 32); sd += __shfl_down(sd, o, 32); }
  if (lane == 0) { as[wid] = ss; ad[wid] = sd; }
}

// ---------- edge passes ----------
__device__ __forceinline__ void edge_sd(const int* __restrict__ ei, int eid, int& s, int& d) {
  if (eid < EE) { s = ei[eid]; d = ei[EE + eid]; } else { s = d = eid - EE; }
}

__global__ void edge_max_kernel(const int* __restrict__ ei,
                                const float* __restrict__ as, const float* __restrict__ ad,
                                unsigned* __restrict__ m, int H) {
  long long t = blockIdx.x * (long long)blockDim.x + threadIdx.x;
  if (t >= (long long)ET * H) return;
  int eid = (int)(t / H), h = (int)(t - (long long)eid * H);
  int s, d; edge_sd(ei, eid, s, d);
  float e = as[(size_t)s * H + h] + ad[(size_t)d * H + h];
  e = e > 0.f ? e : 0.2f * e;
  atomicMax(&m[(size_t)d * H + h], fkey(e));
}

__global__ void edge_expsum_kernel(const int* __restrict__ ei,
                                   const float* __restrict__ as, const float* __restrict__ ad,
                                   const unsigned* __restrict__ m, float* __restrict__ den, int H) {
  long long t = blockIdx.x * (long long)blockDim.x + threadIdx.x;
  if (t >= (long long)ET * H) return;
  int eid = (int)(t / H), h = (int)(t - (long long)eid * H);
  int s, d; edge_sd(ei, eid, s, d);
  float e = as[(size_t)s * H + h] + ad[(size_t)d * H + h];
  e = e > 0.f ? e : 0.2f * e;
  atomicAdd(&den[(size_t)d * H + h], __expf(e - funkey(m[(size_t)d * H + h])));
}

// 4 edges per 256-thread block; 64 threads per edge, float4 per thread (H*C == 256)
__global__ __launch_bounds__(256)
void edge_aggr_kernel(const int* __restrict__ ei,
                      const float* __restrict__ as, const float* __restrict__ ad,
                      const unsigned* __restrict__ m, const float* __restrict__ den,
                      const float* __restrict__ xp, int ldxp,
                      float* __restrict__ out, int H, int C) {
  int eid = blockIdx.x * 4 + (threadIdx.x >> 6);
  int l64 = threadIdx.x & 63;         // 64 threads per edge
  int c0  = l64 * 4;                  // feature base (0..252)
  int h   = c0 / C;
  int s, d; edge_sd(ei, eid, s, d);
  float e = as[(size_t)s * H + h] + ad[(size_t)d * H + h];
  e = e > 0.f ? e : 0.2f * e;
  float alpha = __expf(e - funkey(m[(size_t)d * H + h])) / (den[(size_t)d * H + h] + 1e-16f);
  const float4 xv = *(const float4*)&xp[(size_t)s * ldxp + c0];
  float* op = &out[(size_t)d * 256 + c0];
  atomicAdd(op + 0, xv.x * alpha);
  atomicAdd(op + 1, xv.y * alpha);
  atomicAdd(op + 2, xv.z * alpha);
  atomicAdd(op + 3, xv.w * alpha);
}

// ---------- BatchNorm ----------
__global__ void bn_stats_kernel(const float* __restrict__ out0, const float* __restrict__ b0,
                                float* __restrict__ bsum, float* __restrict__ bsq) {
  int col = threadIdx.x;          // 256 columns
  int r0 = blockIdx.x * 256;
  float b = b0[col];
  float s = 0.f, q = 0.f;
  for (int i = 0; i < 256; ++i) {
    int r = r0 + i;
    if (r < NN) { float v = out0[(size_t)r * 256 + col] + b; s += v; q += v * v; }
  }
  atomicAdd(&bsum[col], s);
  atomicAdd(&bsq[col], q);
}

__global__ void bn_final_kernel(const float* __restrict__ bsum, const float* __restrict__ bsq,
                                float* __restrict__ mu, float* __restrict__ rstd) {
  int j = threadIdx.x;
  float m = bsum[j] * (1.f / NN);
  float v = bsq[j] * (1.f / NN) - m * m;
  mu[j] = m;
  rstd[j] = rsqrtf(v + 1e-5f);
}

__global__ void bn_elu_skip_kernel(const float* __restrict__ out0, const float* __restrict__ xp0skip,
                                   const float* __restrict__ b0, const float* __restrict__ gamma0,
                                   const float* __restrict__ beta0, const float* __restrict__ b_skip,
                                   const float* __restrict__ mu, const float* __restrict__ rstd,
                                   unsigned short* __restrict__ hb) {
  long long t = blockIdx.x * (long long)blockDim.x + threadIdx.x;
  if (t >= (long long)NN * 256) return;
  int n = (int)(t >> 8), j = (int)(t & 255);
  float v = out0[t] + b0[j];
  v = (v - mu[j]) * rstd[j] * gamma0[j] + beta0[j];
  v = v > 0.f ? v : (__expf(v) - 1.f);                // ELU
  v += xp0skip[(size_t)n * 512 + 256 + j] + b_skip[j];
  hb[t] = f2bf(v);
}

__global__ void add_bias_kernel(float* __restrict__ out, const float* __restrict__ b) {
  long long t = blockIdx.x * (long long)blockDim.x + threadIdx.x;
  if (t < (long long)NN * 256) out[t] += b[t & 255];
}

// ---------- host launch ----------
extern "C" void kernel_launch(void* const* d_in, const int* in_sizes, int n_in,
                              void* d_out, int out_size, void* d_ws, size_t ws_size,
                              hipStream_t stream) {
  (void)in_sizes; (void)n_in; (void)out_size; (void)ws_size;
  const float* x      = (const float*)d_in[0];
  const int*   ei     = (const int*)d_in[1];
  const float* W0     = (const float*)d_in[2];
  const float* att_s0 = (const float*)d_in[3];
  const float* att_d0 = (const float*)d_in[4];
  const float* b0     = (const float*)d_in[5];
  const float* gamma0 = (const float*)d_in[6];
  const float* beta0  = (const float*)d_in[7];
  const float* W_skip = (const float*)d_in[8];
  const float* b_skip = (const float*)d_in[9];
  const float* W1     = (const float*)d_in[10];
  const float* att_s1 = (const float*)d_in[11];
  const float* att_d1 = (const float*)d_in[12];
  const float* b1     = (const float*)d_in[13];
  float* out = (float*)d_out;

  char* ws = (char*)d_ws;
  const size_t OFF_XB   = 0;                                    // x bf16   [N,128]
  const size_t OFF_HB   = OFF_XB   + (size_t)NN * 128 * 2;      // h bf16   [N,256]
  const size_t OFF_B0P  = OFF_HB   + (size_t)NN * 256 * 2;      // packed [W0|Wskip] 128x512
  const size_t OFF_B1P  = OFF_B0P  + (size_t)128 * 512 * 2;     // packed W1 256x256
  const size_t OFF_XP0  = OFF_B1P  + (size_t)256 * 256 * 2;     // f32 [N,512] (xp0|skip), reused as xp1 [N,256]
  const size_t OFF_OUT0 = OFF_XP0  + (size_t)NN * 512 * 4;      // f32 [N,256]
  const size_t OFF_AS0  = OFF_OUT0 + (size_t)NN * 256 * 4;
  const size_t OFF_AD0  = OFF_AS0  + (size_t)NN * 8 * 4;
  const size_t OFF_M0   = OFF_AD0  + (size_t)NN * 8 * 4;
  const size_t OFF_DEN0 = OFF_M0   + (size_t)NN * 8 * 4;
  const size_t OFF_AS1  = OFF_DEN0 + (size_t)NN * 8 * 4;
  const size_t OFF_AD1  = OFF_AS1  + (size_t)NN * 4;
  const size_t OFF_M1   = OFF_AD1  + (size_t)NN * 4;
  const size_t OFF_DEN1 = OFF_M1   + (size_t)NN * 4;
  const size_t OFF_BST  = OFF_DEN1 + (size_t)NN * 4;            // 4 x 256 f32: sum, sq, mu, rstd

  unsigned short* xb   = (unsigned short*)(ws + OFF_XB);
  unsigned short* hb   = (unsigned short*)(ws + OFF_HB);
  unsigned short* b0p  = (unsigned short*)(ws + OFF_B0P);
  unsigned short* b1p  = (unsigned short*)(ws + OFF_B1P);
  float* xp0  = (float*)(ws + OFF_XP0);
  float* out0 = (float*)(ws + OFF_OUT0);
  float* as0  = (float*)(ws + OFF_AS0);
  float* ad0  = (float*)(ws + OFF_AD0);
  unsigned* m0 = (unsigned*)(ws + OFF_M0);
  float* den0 = (float*)(ws + OFF_DEN0);
  float* as1  = (float*)(ws + OFF_AS1);
  float* ad1  = (float*)(ws + OFF_AD1);
  unsigned* m1 = (unsigned*)(ws + OFF_M1);
  float* den1 = (float*)(ws + OFF_DEN1);
  float* bsum = (float*)(ws + OFF_BST);
  float* bsq  = bsum + 256;
  float* bmu  = bsum + 512;
  float* brstd = bsum + 768;

  const unsigned KEY_NEG_INF = 0x007FFFFFu;  // fkey(-inf)

  // ---- init / convert / pack ----
  f32_to_bf16_kernel<<<(NN * 128 + 255) / 256, 256, 0, stream>>>(x, xb, (long long)NN * 128);
  pack_b_kernel<<<(128 * 512) / 256, 256, 0, stream>>>(W0, W_skip, 256, 256, b0p, 512);
  pack_b_kernel<<<(256 * 256) / 256, 256, 0, stream>>>(W1, W1, 256, 256, b1p, 256);
  fill_u32_kernel<<<(NN * 256 + 255) / 256, 256, 0, stream>>>((unsigned*)out0, 0u, (long long)NN * 256);
  fill_u32_kernel<<<(NN * 8 + 255) / 256, 256, 0, stream>>>((unsigned*)den0, 0u, (long long)NN * 8);
  fill_u32_kernel<<<(NN * 8 + 255) / 256, 256, 0, stream>>>(m0, KEY_NEG_INF, (long long)NN * 8);
  fill_u32_kernel<<<(NN + 255) / 256, 256, 0, stream>>>((unsigned*)den1, 0u, (long long)NN);
  fill_u32_kernel<<<(NN + 255) / 256, 256, 0, stream>>>(m1, KEY_NEG_INF, (long long)NN);
  fill_u32_kernel<<<(NN * 256 + 255) / 256, 256, 0, stream>>>((unsigned*)out, 0u, (long long)NN * 256);
  fill_u32_kernel<<<2, 256, 0, stream>>>((unsigned*)bsum, 0u, 512);

  // ---- layer 0: fused GEMM x@[W0|Wskip] -> [N,512] ----
  gemm_bf16_wmma_kernel<128><<<dim3(NN / 64, 512 / 64), 128, 0, stream>>>(xb, b0p, xp0, 512);
  att_scores_kernel<<<(NN * 8 * 32 + 255) / 256, 256, 0, stream>>>(xp0, 512, att_s0, att_d0,
                                                                   as0, ad0, NN * 8, 8, 32);
  edge_max_kernel<<<((long long)ET * 8 + 255) / 256, 256, 0, stream>>>(ei, as0, ad0, m0, 8);
  edge_expsum_kernel<<<((long long)ET * 8 + 255) / 256, 256, 0, stream>>>(ei, as0, ad0, m0, den0, 8);
  edge_aggr_kernel<<<ET / 4, 256, 0, stream>>>(ei, as0, ad0, m0, den0, xp0, 512, out0, 8, 32);

  // ---- BN + ELU + skip -> bf16 h ----
  bn_stats_kernel<<<(NN + 255) / 256, 256, 0, stream>>>(out0, b0, bsum, bsq);
  bn_final_kernel<<<1, 256, 0, stream>>>(bsum, bsq, bmu, brstd);
  bn_elu_skip_kernel<<<(NN * 256 + 255) / 256, 256, 0, stream>>>(out0, xp0, b0, gamma0, beta0,
                                                                 b_skip, bmu, brstd, hb);

  // ---- layer 1: GEMM h@W1 -> xp1 (reuses xp0 buffer as [N,256]) ----
  gemm_bf16_wmma_kernel<256><<<dim3(NN / 64, 256 / 64), 128, 0, stream>>>(hb, b1p, xp0, 256);
  att_scores_kernel<<<(NN * 32 + 255) / 256, 256, 0, stream>>>(xp0, 256, att_s1, att_d1,
                                                               as1, ad1, NN, 1, 256);
  edge_max_kernel<<<(ET + 255) / 256, 256, 0, stream>>>(ei, as1, ad1, m1, 1);
  edge_expsum_kernel<<<(ET + 255) / 256, 256, 0, stream>>>(ei, as1, ad1, m1, den1, 1);
  edge_aggr_kernel<<<ET / 4, 256, 0, stream>>>(ei, as1, ad1, m1, den1, xp0, 256, out, 1, 256);
  add_bias_kernel<<<(NN * 256 + 255) / 256, 256, 0, stream>>>(out, b1);
}
